// SpectralGNN_36369783062878
// MI455X (gfx1250) — matrix operand order repaired and checked by reading, not compiled, not run on hardware
//
#include <hip/hip_runtime.h>

// ---------------- problem constants (match reference) ----------------
#define N_NODES  20000
#define N_PAD    20096      // rows padded to multiple of BM for guard-free tile copies
#define N_EDGES  320000
#define D_IN_    800
#define D_H_     256
#define D_OUT_   800
#define D_OUTP_  896        // W_out cols padded to multiple of BN
#define NLAYERS_ 3
#define NEG_SLOPE_ 0.2f
#define BN_EPS_  1e-5f

typedef unsigned short u16;

// ---------------- WMMA / vector types ----------------
typedef __attribute__((ext_vector_type(16))) __bf16 v16bf;
typedef __attribute__((ext_vector_type(8)))  __bf16 bf16x8;
typedef __attribute__((ext_vector_type(8)))  float  v8f;
typedef __attribute__((ext_vector_type(4)))  int    v4i;

// ---------------- GEMM tiling ----------------
#define BM 128
#define BN 128
#define BK 32
#define ASTR 40   // LDS row stride in bf16 elems (80B -> keeps 16B alignment)
#define BSTR 40

#if __has_builtin(__builtin_amdgcn_global_load_async_to_lds_b128)
#define HAVE_ASYNC_LDS 1
#else
#define HAVE_ASYNC_LDS 0
#endif

__device__ inline void wait_async_lds() {
#if HAVE_ASYNC_LDS
#if __has_builtin(__builtin_amdgcn_s_wait_asynccnt)
    __builtin_amdgcn_s_wait_asynccnt(0);
#else
    asm volatile("s_wait_asynccnt 0x0" ::: "memory");
#endif
#endif
}

__device__ inline u16 f2bf_bits(float f) {
    union { float f; unsigned u; } a; a.f = f;
    return (u16)((a.u + 0x7FFFu + ((a.u >> 16) & 1u)) >> 16);  // RNE
}

// C[M,N](f32) = A[M,K](bf16) @ B[K,N](bf16), f32 accumulate via WMMA.
// A rows and B cols may be padded (garbage beyond M/N never stored).
// Epilogue: + bias[n], + residual[m*ldr + n].
__global__ __launch_bounds__(256)
void gat_gemm_bf16(const u16* __restrict__ A, int lda,
                   const u16* __restrict__ B, int ldb,
                   float* __restrict__ C, int ldc,
                   int M, int N, int K,
                   const float* __restrict__ bias,
                   const float* __restrict__ residual, int ldr)
{
    __shared__ __bf16 ldsA[BM * ASTR];   // row-major A tile
    __shared__ __bf16 ldsBt[BN * BSTR];  // B tile transposed: Bt[n][k]

    const int tid  = threadIdx.x;
    const int lane = tid & 31;
    const int wave = tid >> 5;       // 8 waves
    const int wm   = wave & 3;       // wave M slot: 32 rows each
    const int wn   = wave >> 2;      // wave N slot: 64 cols each
    const int rowBase = blockIdx.x * BM;
    const int colBase = blockIdx.y * BN;
    const int l16 = lane & 15;
    const int lhi = lane >> 4;

    v8f acc[2][4] = {};

    for (int k0 = 0; k0 < K; k0 += BK) {
        // ---- A tile: 128x32 bf16 = 512 x 16B chunks, 2 per thread ----
        #pragma unroll
        for (int it = 0; it < 2; ++it) {
            int c = tid + it * 256;
            int r = c >> 2;                   // row 0..127
            int seg = c & 3;                  // 8 bf16 per chunk
            const u16* gp = A + (size_t)(rowBase + r) * lda + k0 + seg * 8;
            __bf16* lp = &ldsA[r * ASTR + seg * 8];
#if HAVE_ASYNC_LDS
            __builtin_amdgcn_global_load_async_to_lds_b128((v4i*)gp, (v4i*)lp, 0, 0);
#else
            *(bf16x8*)lp = *(const bf16x8*)gp;
#endif
        }
        // ---- B tile: 32x128 bf16, transpose into LDS (2 chunks/thread) ----
        #pragma unroll
        for (int it = 0; it < 2; ++it) {
            int c = tid + it * 256;
            int kk = c >> 4;                  // 0..31
            int seg = c & 15;                 // cols seg*8..seg*8+7
            const u16* gp = B + (size_t)(k0 + kk) * ldb + colBase + seg * 8;
            bf16x8 v = *(const bf16x8*)gp;
            #pragma unroll
            for (int j = 0; j < 8; ++j)
                ldsBt[(seg * 8 + j) * BSTR + kk] = v[j];
        }
        wait_async_lds();
        __syncthreads();

        // ---- fragments per ISA 16-bit layouts ----
        // A 16x32: lanes 0-15 K{0..7,16..23}; lanes 16-31 K{8..15,24..31}
        union FR { v16bf v; bf16x8 h[2]; };
        FR a[2], b[4];
        #pragma unroll
        for (int mi = 0; mi < 2; ++mi) {
            int ar = wm * 32 + mi * 16 + l16;
            int ko = lhi * 8;
            a[mi].h[0] = *(const bf16x8*)&ldsA[ar * ASTR + ko];
            a[mi].h[1] = *(const bf16x8*)&ldsA[ar * ASTR + ko + 16];
        }
        // B 32x16: lane%16 = col n; lanes 0-15 K{0..7,16..23}; lanes 16-31 K{8..15,24..31}
        #pragma unroll
        for (int ni = 0; ni < 4; ++ni) {
            int bc = wn * 64 + ni * 16 + l16;
            int kb = lhi * 8;
            b[ni].h[0] = *(const bf16x8*)&ldsBt[bc * BSTR + kb];
            b[ni].h[1] = *(const bf16x8*)&ldsBt[bc * BSTR + kb + 16];
        }
        #pragma unroll
        for (int mi = 0; mi < 2; ++mi)
            #pragma unroll
            for (int ni = 0; ni < 4; ++ni)
                acc[mi][ni] = __builtin_amdgcn_wmma_f32_16x16x32_bf16(
                    false, a[mi].v, false, b[ni].v, (short)0, acc[mi][ni],
                    false, false);
        __syncthreads();
    }

    // ---- epilogue: lane(l16)=col, lhi*8+r = row within 16x16 tile ----
    #pragma unroll
    for (int mi = 0; mi < 2; ++mi) {
        #pragma unroll
        for (int ni = 0; ni < 4; ++ni) {
            int col = colBase + wn * 64 + ni * 16 + l16;
            if (col >= N) continue;
            float bv = bias ? bias[col] : 0.f;
            int rbase = rowBase + wm * 32 + mi * 16 + lhi * 8;
            #pragma unroll
            for (int r = 0; r < 8; ++r) {
                int row = rbase + r;
                if (row < M) {
                    float v = acc[mi][ni][r] + bv;
                    if (residual) v += residual[(size_t)row * ldr + col];
                    C[(size_t)row * ldc + col] = v;
                }
            }
        }
    }
}

// ---------------- conversion kernels ----------------
__global__ void f32_to_bf16(const float* __restrict__ in, u16* __restrict__ out, int n) {
    int i = blockIdx.x * blockDim.x + threadIdx.x;
    if (i < n) out[i] = f2bf_bits(in[i]);
}
__global__ void f32_to_bf16_pad(const float* __restrict__ in, int ldin,
                                u16* __restrict__ out, int ldout, int rows, int cols) {
    int i = blockIdx.x * blockDim.x + threadIdx.x;
    if (i >= rows * ldout) return;
    int r = i / ldout, c = i - r * ldout;
    out[i] = (c < cols) ? f2bf_bits(in[(size_t)r * ldin + c]) : (u16)0;
}

// ---------------- utility fills ----------------
__global__ void fill_f32(float* p, float v, int n) {
    int i = blockIdx.x * blockDim.x + threadIdx.x;
    if (i < n) p[i] = v;
}
__global__ void fill_u32(unsigned* p, unsigned v, int n) {
    int i = blockIdx.x * blockDim.x + threadIdx.x;
    if (i < n) p[i] = v;
}

// ---------------- attention vectors: per-row dual dot product ----------------
__global__ __launch_bounds__(256)
void dot2_kernel(const float* __restrict__ hw,
                 const float* __restrict__ a1, const float* __restrict__ a2,
                 float* __restrict__ o1, float* __restrict__ o2, int n)
{
    int w = (blockIdx.x * blockDim.x + threadIdx.x) >> 5;
    int lane = threadIdx.x & 31;
    if (w >= n) return;
    const float4* hp = (const float4*)(hw + (size_t)w * D_H_);
    const float4* p1 = (const float4*)a1;
    const float4* p2 = (const float4*)a2;
    float s1 = 0.f, s2 = 0.f;
    #pragma unroll
    for (int j = 0; j < 2; ++j) {
        float4 h4 = hp[lane * 2 + j];
        float4 w1 = p1[lane * 2 + j];
        float4 w2 = p2[lane * 2 + j];
        s1 += h4.x * w1.x + h4.y * w1.y + h4.z * w1.z + h4.w * w1.w;
        s2 += h4.x * w2.x + h4.y * w2.y + h4.z * w2.z + h4.w * w2.w;
    }
    #pragma unroll
    for (int o = 16; o > 0; o >>= 1) {
        s1 += __shfl_xor(s1, o, 32);
        s2 += __shfl_xor(s2, o, 32);
    }
    if (lane == 0) { o1[w] = s1; o2[w] = s2; }
}

// ---------------- ordered-uint float max trick ----------------
__device__ inline unsigned f2ord(float f) {
    unsigned u = __float_as_uint(f);
    return (u & 0x80000000u) ? ~u : (u | 0x80000000u);
}
__device__ inline float ord2f(unsigned u) {
    return (u & 0x80000000u) ? __uint_as_float(u & 0x7FFFFFFFu)
                             : __uint_as_float(~u);
}

// ---------------- edge kernels (segment softmax + scatter) ----------------
__global__ void edge_logits_max(const float* __restrict__ asv, const float* __restrict__ adv,
                                const int* __restrict__ src, const int* __restrict__ dst,
                                float* __restrict__ e, unsigned* __restrict__ mord, int ne)
{
    int j = blockIdx.x * blockDim.x + threadIdx.x;
    if (j >= ne) return;
    float v = asv[src[j]] + adv[dst[j]];
    v = v > 0.f ? v : NEG_SLOPE_ * v;
    e[j] = v;
    atomicMax(&mord[dst[j]], f2ord(v));
}

__global__ void edge_exp_sum(float* __restrict__ e, const unsigned* __restrict__ mord,
                             const int* __restrict__ dst, float* __restrict__ z, int ne)
{
    int j = blockIdx.x * blockDim.x + threadIdx.x;
    if (j >= ne) return;
    int d = dst[j];
    float p = __expf(e[j] - ord2f(mord[d]));
    e[j] = p;
    atomicAdd(&z[d], p);
}

// one wave32 per edge: coalesced 256-float gather + atomic scatter
__global__ __launch_bounds__(256)
void edge_scatter(const float* __restrict__ p, const float* __restrict__ z,
                  const int* __restrict__ src, const int* __restrict__ dst,
                  const float* __restrict__ hw, float* __restrict__ agg, int ne)
{
    int w = (blockIdx.x * 256 + threadIdx.x) >> 5;
    int lane = threadIdx.x & 31;
    if (w >= ne) return;
    int s = src[w], d = dst[w];
    float c = p[w] / z[d];
    const float4* hs = (const float4*)(hw + (size_t)s * D_H_);
    float* ap = agg + (size_t)d * D_H_;
    #pragma unroll
    for (int j = 0; j < 2; ++j) {
        float4 v = hs[lane * 2 + j];
        int base = (lane * 2 + j) * 4;
        atomicAdd(ap + base + 0, c * v.x);
        atomicAdd(ap + base + 1, c * v.y);
        atomicAdd(ap + base + 2, c * v.z);
        atomicAdd(ap + base + 3, c * v.w);
    }
}

// ---------------- BatchNorm (training-mode batch stats) ----------------
__global__ __launch_bounds__(256)
void bn_stats(const float* __restrict__ in, const float* __restrict__ prebias,
              float* __restrict__ sum_acc, float* __restrict__ sq_acc,
              int rows, int rowsPerBlock)
{
    int col = threadIdx.x;
    int r0 = blockIdx.x * rowsPerBlock;
    int r1 = r0 + rowsPerBlock; if (r1 > rows) r1 = rows;
    float pb = prebias ? prebias[col] : 0.f;
    float s = 0.f, ss = 0.f;
    for (int r = r0; r < r1; ++r) {
        float v = in[(size_t)r * D_H_ + col] + pb;
        s += v; ss += v * v;
    }
    atomicAdd(&sum_acc[col], s);
    atomicAdd(&sq_acc[col], ss);
}

__global__ void bn_finalize(const float* __restrict__ sum_acc, const float* __restrict__ sq_acc,
                            float* __restrict__ mean, float* __restrict__ rstd, int rows)
{
    int c = threadIdx.x;
    float mu = sum_acc[c] / (float)rows;
    float var = sq_acc[c] / (float)rows - mu * mu;
    mean[c] = mu;
    rstd[c] = rsqrtf(var + BN_EPS_);
}

// BN apply, optional relu / residual; writes f32 h and bf16 h for next GEMM
__global__ void bn_apply(const float* __restrict__ in, const float* __restrict__ prebias,
                         const float* __restrict__ mean, const float* __restrict__ rstd,
                         const float* __restrict__ g, const float* __restrict__ be,
                         const float* __restrict__ residual,
                         float* __restrict__ out, u16* __restrict__ out_bf,
                         int n_elem, int do_relu)
{
    int i = blockIdx.x * blockDim.x + threadIdx.x;
    if (i >= n_elem) return;
    int c = i & (D_H_ - 1);
    float v = in[i] + (prebias ? prebias[c] : 0.f);
    v = g[c] * (v - mean[c]) * rstd[c] + be[c];
    if (do_relu) v = fmaxf(v, 0.f);
    if (residual) v += residual[i];
    out[i] = v;
    if (out_bf) out_bf[i] = f2bf_bits(v);
}

// ---------------- orchestration ----------------
extern "C" void kernel_launch(void* const* d_in, const int* in_sizes, int n_in,
                              void* d_out, int out_size, void* d_ws, size_t ws_size,
                              hipStream_t stream) {
    (void)in_sizes; (void)n_in; (void)out_size; (void)ws_size;
    const float* x      = (const float*)d_in[0];
    const int*   ei     = (const int*)d_in[1];
    const int*   src    = ei;
    const int*   dst    = ei + N_EDGES;
    const float* b_in   = (const float*)d_in[3];
    const float* g_in   = (const float*)d_in[4];
    const float* be_in  = (const float*)d_in[5];
    const float* att_s  = (const float*)d_in[7];
    const float* att_d  = (const float*)d_in[8];
    const float* bias_l = (const float*)d_in[9];
    const float* g_l    = (const float*)d_in[10];
    const float* be_l   = (const float*)d_in[11];
    const float* b_out  = (const float*)d_in[13];
    float* out = (float*)d_out;

    float* ws = (float*)d_ws;
    size_t off = 0;
    float*    h       = ws + off; off += (size_t)N_NODES * D_H_;
    float*    h_prev  = ws + off; off += (size_t)N_NODES * D_H_;
    float*    hw      = ws + off; off += (size_t)N_NODES * D_H_;
    float*    agg     = ws + off; off += (size_t)N_NODES * D_H_;
    float*    asv     = ws + off; off += N_NODES;
    float*    adv     = ws + off; off += N_NODES;
    unsigned* mord    = (unsigned*)(ws + off); off += N_NODES;
    float*    zbuf    = ws + off; off += N_NODES;
    float*    sum_acc = ws + off; off += D_H_;   // sum_acc & sq_acc contiguous
    float*    sq_acc  = ws + off; off += D_H_;
    float*    mean    = ws + off; off += D_H_;
    float*    rstd    = ws + off; off += D_H_;
    float*    ebuf    = ws + off; off += N_EDGES;
    u16*      xbf     = (u16*)(ws + off); off += (size_t)N_PAD * D_IN_ / 2;
    u16*      hbf     = (u16*)(ws + off); off += (size_t)N_PAD * D_H_ / 2;
    u16*      Winbf   = (u16*)(ws + off); off += (size_t)D_IN_ * D_H_ / 2;
    u16*      Wlbf    = (u16*)(ws + off); off += (size_t)NLAYERS_ * D_H_ * D_H_ / 2;
    u16*      Woutbf  = (u16*)(ws + off); off += (size_t)D_H_ * D_OUTP_ / 2;

    const dim3 blk(256);
    auto cdiv = [](int a, int b) { return (a + b - 1) / b; };

    auto gemm = [&](const u16* A, int lda, const u16* B, int ldb,
                    float* C, int ldc, int M, int Nn, int K,
                    const float* bias, const float* res, int ldr) {
        dim3 grid(cdiv(M, BM), cdiv(Nn, BN));
        gat_gemm_bf16<<<grid, blk, 0, stream>>>(A, lda, B, ldb, C, ldc,
                                                M, Nn, K, bias, res, ldr);
    };
    auto run_bn = [&](const float* in, const float* pb, const float* g,
                      const float* be, const float* res, float* o, u16* obf, int relu) {
        fill_f32<<<cdiv(2 * D_H_, 256), blk, 0, stream>>>(sum_acc, 0.f, 2 * D_H_);
        bn_stats<<<80, blk, 0, stream>>>(in, pb, sum_acc, sq_acc, N_NODES, 250);
        bn_finalize<<<1, blk, 0, stream>>>(sum_acc, sq_acc, mean, rstd, N_NODES);
        int ne = N_NODES * D_H_;
        bn_apply<<<cdiv(ne, 256), blk, 0, stream>>>(in, pb, mean, rstd, g, be, res, o, obf, ne, relu);
    };

    // 0) precompute bf16 operands (x, weights); pads never influence stored C
    f32_to_bf16<<<cdiv(N_NODES * D_IN_, 256), blk, 0, stream>>>(
        (const float*)d_in[0], xbf, N_NODES * D_IN_);
    f32_to_bf16<<<cdiv(D_IN_ * D_H_, 256), blk, 0, stream>>>(
        (const float*)d_in[2], Winbf, D_IN_ * D_H_);
    f32_to_bf16<<<cdiv(NLAYERS_ * D_H_ * D_H_, 256), blk, 0, stream>>>(
        (const float*)d_in[6], Wlbf, NLAYERS_ * D_H_ * D_H_);
    f32_to_bf16_pad<<<cdiv(D_H_ * D_OUTP_, 256), blk, 0, stream>>>(
        (const float*)d_in[12], D_OUT_, Woutbf, D_OUTP_, D_H_, D_OUT_);

    // 1) input projection + BN + ReLU
    gemm(xbf, D_IN_, Winbf, D_H_, hw, D_H_, N_NODES, D_H_, D_IN_, b_in, nullptr, 0);
    run_bn(hw, nullptr, g_in, be_in, nullptr, h, hbf, 1);

    // 2) GAT layers
    for (int i = 0; i < NLAYERS_; ++i) {
        if (i == 1)
            (void)hipMemcpyAsync(h_prev, h, (size_t)N_NODES * D_H_ * sizeof(float),
                                 hipMemcpyDeviceToDevice, stream);
        gemm(hbf, D_H_, Wlbf + (size_t)i * D_H_ * D_H_, D_H_, hw, D_H_,
             N_NODES, D_H_, D_H_, nullptr, nullptr, 0);
        dot2_kernel<<<cdiv(N_NODES, 8), blk, 0, stream>>>(
            hw, att_s + (size_t)i * D_H_, att_d + (size_t)i * D_H_, asv, adv, N_NODES);

        fill_u32<<<cdiv(N_NODES, 256), blk, 0, stream>>>(mord, 0x007FFFFFu, N_NODES); // ord(-inf)
        fill_f32<<<cdiv(N_NODES, 256), blk, 0, stream>>>(zbuf, 0.f, N_NODES);
        fill_f32<<<cdiv(N_NODES * D_H_, 256), blk, 0, stream>>>(agg, 0.f, N_NODES * D_H_);

        edge_logits_max<<<cdiv(N_EDGES, 256), blk, 0, stream>>>(asv, adv, src, dst, ebuf, mord, N_EDGES);
        edge_exp_sum<<<cdiv(N_EDGES, 256), blk, 0, stream>>>(ebuf, mord, dst, zbuf, N_EDGES);
        edge_scatter<<<cdiv(N_EDGES, 8), blk, 0, stream>>>(ebuf, zbuf, src, dst, hw, agg, N_EDGES);

        run_bn(agg, bias_l + (size_t)i * D_H_, g_l + (size_t)i * D_H_,
               be_l + (size_t)i * D_H_,
               (i == 1) ? h_prev : nullptr, h, hbf, (i < NLAYERS_ - 1) ? 1 : 0);
    }

    // 3) output projection + b_out + input residual
    gemm(hbf, D_H_, Woutbf, D_OUTP_, out, D_OUT_, N_NODES, D_OUT_, D_H_, b_out, x, D_IN_);
}
